// MultiHeadedAttention_84361747628298
// MI455X (gfx1250) — matrix operand order repaired and checked by reading, not compiled, not run on hardware
//
#include <hip/hip_runtime.h>
#include <hip/hip_bf16.h>

typedef _Float16 half_t;
typedef __attribute__((ext_vector_type(16))) _Float16 v16h;
typedef __attribute__((ext_vector_type(8)))  _Float16 v8h;
typedef __attribute__((ext_vector_type(4)))  _Float16 v4h;
typedef __attribute__((ext_vector_type(8)))  float    v8f;
typedef __attribute__((ext_vector_type(4)))  float    v4f;
typedef int v4i __attribute__((vector_size(16)));

union f16x16 { v16h v; v8h h8[2]; };

#define B_  4
#define S_  1024
#define D_  512
#define H_  8
#define E_  64
#define BS  (B_ * S_)          // 4096 rows

// ---- CDNA5 async global->LDS copy (ASYNCcnt path), guarded ----------------
#if defined(__has_builtin)
#  if __has_builtin(__builtin_amdgcn_global_load_async_to_lds_b128)
#    define HAVE_ASYNC_LDS 1
#  endif
#  if __has_builtin(__builtin_amdgcn_s_wait_asynccnt)
#    define HAVE_WAIT_ASYNC 1
#  endif
#endif

__device__ __forceinline__ void async_copy16(const half_t* g, half_t* l) {
#ifdef HAVE_ASYNC_LDS
    typedef __attribute__((address_space(1))) v4i* gp_t;
    typedef __attribute__((address_space(3))) v4i* lp_t;
    v4i* gg = (v4i*)(void*)(half_t*)g;   // strip const, generic AS
    v4i* ll = (v4i*)(void*)l;
    __builtin_amdgcn_global_load_async_to_lds_b128((gp_t)gg, (lp_t)ll, 0, 0);
#else
    *(v8h*)l = *(const v8h*)g;
#endif
}
__device__ __forceinline__ void async_wait() {
#ifdef HAVE_ASYNC_LDS
#  ifdef HAVE_WAIT_ASYNC
    __builtin_amdgcn_s_wait_asynccnt(0);
#  else
    asm volatile("s_wait_asynccnt 0x0" ::: "memory");
#  endif
#endif
}

// ---------------------------------------------------------------------------
// Kernel 0: Wf_sum[e][d] = sum_h Wf[h*E + e][d]
// ---------------------------------------------------------------------------
__global__ __launch_bounds__(256) void wfsum_kernel(const float* __restrict__ Wf,
                                                    float* __restrict__ Wfs) {
    int i = blockIdx.x * 256 + threadIdx.x;          // 64*512 = 32768 elems
    if (i < E_ * D_) {
        int e = i / D_, d = i % D_;
        float s = 0.f;
#pragma unroll
        for (int h = 0; h < H_; ++h)
            s += Wf[(size_t)(h * E_ + e) * D_ + d];
        Wfs[i] = s;
    }
}

// ---------------------------------------------------------------------------
// Kernel 1: projections. 17 GEMM slices of [4096x512] x [512x64].
//   slice 0..7  : Q head h -> Qh[h]  row-major f16 [B*S][E]
//   slice 8..15 : K head h -> KhT[h] TRANSPOSED f16 [B][E][S]
//   slice 16    : V shared -> Vh     row-major f16 [B*S][E]
// ---------------------------------------------------------------------------
__global__ __launch_bounds__(256) void proj_kernel(
    const float* __restrict__ query, const float* __restrict__ key,
    const float* __restrict__ value,
    const float* __restrict__ Wq, const float* __restrict__ bq,
    const float* __restrict__ Wk, const float* __restrict__ bk,
    const float* __restrict__ Wv, const float* __restrict__ bv,
    half_t* __restrict__ Qh, half_t* __restrict__ KhT, half_t* __restrict__ Vh)
{
    __shared__ alignas(16) half_t As[2][128 * 32];   // [row][k]  row stride 64B
    __shared__ alignas(16) half_t Bs[2][32 * 64];    // [k][n]    row stride 128B

    const int slice = blockIdx.y;
    const int r0    = blockIdx.x * 128;

    const float* A; const float* W; const float* bias; half_t* Out;
    int is_k = 0;
    if (slice < 8)        { A = query; W = Wq + (size_t)slice * D_ * E_; bias = bq + slice * E_; Out = Qh + (size_t)slice * BS * E_; }
    else if (slice < 16)  { int h = slice - 8; is_k = 1;
                            A = key;   W = Wk + (size_t)h * D_ * E_;     bias = bk + h * E_;     Out = KhT + (size_t)h * B_ * E_ * S_; }
    else                  { A = value; W = Wv;                           bias = bv;              Out = Vh; }

    const int t    = threadIdx.x;
    const int wave = t >> 5, lane = t & 31;
    const int hf   = lane >> 4, lr = lane & 15;
    const int m0   = r0 + wave * 16;

    // per-thread staging coordinates
    const int arow[4] = { (t + 0)   >> 3, (t + 256) >> 3, (t + 512) >> 3, (t + 768) >> 3 };
    const int acg     = (t & 7) * 4;
    const int brow[2] = { (t + 0) >> 4, (t + 256) >> 4 };
    const int bcg     = (t & 15) * 4;

    v8f acc[4] = {};

    int kstep = 0;
    for (int k0 = 0; k0 < D_; k0 += 32, ++kstep) {
        const int buf = kstep & 1;
        // phase 1: issue all staging loads (one clause, one wait)
        v4f a_[4], w_[2];
#pragma unroll
        for (int j = 0; j < 4; ++j)
            a_[j] = *(const v4f*)(A + (size_t)(r0 + arow[j]) * D_ + k0 + acg);
#pragma unroll
        for (int j = 0; j < 2; ++j)
            w_[j] = *(const v4f*)(W + (size_t)(k0 + brow[j]) * E_ + bcg);
        // phase 2: convert + store to LDS
#pragma unroll
        for (int j = 0; j < 4; ++j) {
            v4h ah; ah[0] = (half_t)a_[j][0]; ah[1] = (half_t)a_[j][1];
                    ah[2] = (half_t)a_[j][2]; ah[3] = (half_t)a_[j][3];
            *(v4h*)(&As[buf][arow[j] * 32 + acg]) = ah;
        }
#pragma unroll
        for (int j = 0; j < 2; ++j) {
            v4h wh; wh[0] = (half_t)w_[j][0]; wh[1] = (half_t)w_[j][1];
                    wh[2] = (half_t)w_[j][2]; wh[3] = (half_t)w_[j][3];
            *(v4h*)(&Bs[buf][brow[j] * 64 + bcg]) = wh;
        }
        if (k0 + 32 < D_) {   // hint next K-step into cache
            __builtin_prefetch(A + (size_t)(r0 + arow[0]) * D_ + k0 + 32 + acg, 0, 1);
            if (t < 128)
                __builtin_prefetch(W + (size_t)(k0 + 32 + brow[0]) * E_ + bcg, 0, 1);
        }
        __syncthreads();   // double-buffered: single barrier per step

        // pre-load all fragments, then run the WMMA chain
        f16x16 af, bfr[4];
        {
            const v8h* ap = (const v8h*)(&As[buf][(16 * wave + lr) * 32]);
            af.h8[0] = ap[hf];        // K =  8*hf .. +7
            af.h8[1] = ap[2 + hf];    // K = 16+8*hf .. +7
        }
#pragma unroll
        for (int nt = 0; nt < 4; ++nt) {
            const half_t* bp = &Bs[buf][nt * 16 + lr]; // col = lr, elements over K (table order)
            for (int s = 0; s < 16; ++s) bfr[nt].v[s] = bp[(s < 8 ? 8 * hf + s : 8 + 8 * hf + s) * 64];
        }
#pragma unroll
        for (int nt = 0; nt < 4; ++nt)
            acc[nt] = __builtin_amdgcn_wmma_f32_16x16x32_f16(
                false, af.v, false, bfr[nt].v, (short)0, acc[nt], false, false);
    }

    // store f16 with bias; C layout: vgpr k -> row m0+k+8*hf, col = nt*16+lr
#pragma unroll
    for (int nt = 0; nt < 4; ++nt) {
        float bb = bias[nt * 16 + lr];
#pragma unroll
        for (int k = 0; k < 8; ++k) {
            int row = m0 + k + 8 * hf;
            half_t val = (half_t)(acc[nt][k] + bb);
            if (!is_k) {
                Out[(size_t)row * E_ + nt * 16 + lr] = val;
            } else {
                int bb2 = row >> 10, ss = row & 1023;   // [b][e][s]
                Out[((size_t)bb2 * E_ + nt * 16 + lr) * S_ + ss] = val;
            }
        }
    }
}

// ---------------------------------------------------------------------------
// Kernel 2: causal flash attention per (h, b, 128-row q-block).
// 8 waves x 16 q-rows, kv streamed in 32-row tiles.
// K^T and V tiles staged with GLOBAL_LOAD_ASYNC_TO_LDS_B128 (double-buffered).
// ---------------------------------------------------------------------------
__global__ __launch_bounds__(256) void attn_kernel(
    const half_t* __restrict__ Qh, const half_t* __restrict__ KhT,
    const half_t* __restrict__ Vh, float* __restrict__ Oh)
{
    __shared__ alignas(16) half_t Kt[2][64 * 32];  // [e][kv]  row 64B (pre-transposed)
    __shared__ alignas(16) half_t Vs[2][32 * 64];  // [kv][e]  row 128B
    __shared__ alignas(16) half_t Ps[8][16 * 32];  // per-wave P round-trip

    const int h = blockIdx.z, b = blockIdx.y;
    const int qb = blockIdx.x * 128;
    const int t = threadIdx.x, wave = t >> 5, lane = t & 31;
    const int hf = lane >> 4, lr = lane & 15;
    const int q0 = qb + wave * 16;

    const half_t* Qbase  = Qh  + (size_t)(h * B_ + b) * S_ * E_;
    const half_t* KTbase = KhT + (size_t)(h * B_ + b) * E_ * S_;   // [e][s]
    const half_t* Vbase  = Vh  + (size_t)b * S_ * E_;

    // per-thread staging coordinates (16B granules)
    const int e_row = t >> 2;            // 0..63  (K^T row)
    const int n8    = (t & 3) * 8;       // kv sub-offset

    // Q fragments for both 32-wide e-chunks, loaded once
    f16x16 qf[2];
    {
        const v8h* qp = (const v8h*)(Qbase + (size_t)(q0 + lr) * E_);
#pragma unroll
        for (int c = 0; c < 2; ++c) {
            qf[c].h8[0] = qp[4 * c + hf];
            qf[c].h8[1] = qp[4 * c + 2 + hf];
        }
    }

    v8f o[4] = {};
    float m_r[8], l_r[8];
#pragma unroll
    for (int k = 0; k < 8; ++k) { m_r[k] = -1e30f; l_r[k] = 0.f; }

    const int ntiles = (qb + 128) / 32;     // causal upper bound for this block
    for (int it = 0; it < ntiles; ++it) {
        const int kv0 = it * 32;
        const int buf = it & 1;
        // async staging: K^T tile rows are contiguous in KhT; V rows contiguous
        async_copy16(KTbase + (size_t)e_row * S_ + kv0 + n8, &Kt[buf][e_row * 32 + n8]);
        async_copy16(Vbase + (size_t)kv0 * E_ + t * 8,       &Vs[buf][t * 8]);
        if (it + 1 < ntiles) {
            __builtin_prefetch(KTbase + (size_t)e_row * S_ + kv0 + 32 + n8, 0, 1);
            __builtin_prefetch(Vbase + (size_t)(kv0 + 32) * E_ + t * 8, 0, 1);
        }
        async_wait();
        __syncthreads();   // double-buffer => single barrier per tile is safe

        if (kv0 <= q0 + 15) {
            // pre-load all K^T fragments (die before V-fragments are loaded)
            f16x16 kf[4];   // [c*2+nt]
#pragma unroll
            for (int c = 0; c < 2; ++c)
#pragma unroll
                for (int nt = 0; nt < 2; ++nt) {
                    const half_t* bp = &Kt[buf][(32 * c) * 32 + nt * 16 + lr];
                    for (int s = 0; s < 16; ++s)
                        kf[c * 2 + nt].v[s] = bp[(s < 8 ? 8 * hf + s : 8 + 8 * hf + s) * 32];
                }
            // scores S = Q K^T
            v8f s[2] = {};
#pragma unroll
            for (int c = 0; c < 2; ++c)
#pragma unroll
                for (int nt = 0; nt < 2; ++nt)
                    s[nt] = __builtin_amdgcn_wmma_f32_16x16x32_f16(
                        false, qf[c].v, false, kf[c * 2 + nt].v, (short)0, s[nt], false, false);

            // scale + causal mask (matches reference: scale first, then -1e9)
            const float scale = 0.125f; // 1/sqrt(64)
#pragma unroll
            for (int nt = 0; nt < 2; ++nt)
#pragma unroll
                for (int k = 0; k < 8; ++k) {
                    int qrow = q0 + k + 8 * hf;
                    int kvc  = kv0 + nt * 16 + lr;
                    float v  = s[nt][k] * scale;
                    s[nt][k] = (kvc <= qrow) ? v : -1e9f;
                }
            // online softmax: row max / sum across the 16-lane half-group
            float mnew[8], lad[8];
#pragma unroll
            for (int k = 0; k < 8; ++k) {
                float mx = fmaxf(s[0][k], s[1][k]);
#pragma unroll
                for (int sh = 1; sh < 16; sh <<= 1)
                    mx = fmaxf(mx, __shfl_xor(mx, sh, 32));
                mnew[k] = fmaxf(m_r[k], mx);
            }
#pragma unroll
            for (int k = 0; k < 8; ++k) {
                float e0 = __expf(s[0][k] - mnew[k]);
                float e1 = __expf(s[1][k] - mnew[k]);
                s[0][k] = e0; s[1][k] = e1;
                float sm = e0 + e1;
#pragma unroll
                for (int sh = 1; sh < 16; sh <<= 1)
                    sm += __shfl_xor(sm, sh, 32);
                lad[k] = sm;
            }
#pragma unroll
            for (int k = 0; k < 8; ++k) {
                float sc = __expf(m_r[k] - mnew[k]);
                l_r[k] = l_r[k] * sc + lad[k];
                m_r[k] = mnew[k];
#pragma unroll
                for (int nt = 0; nt < 4; ++nt) o[nt][k] *= sc;
            }
            // P (C-layout f32) -> LDS f16 -> A-fragment (in-order LDS, same wave)
            half_t* pw = Ps[wave];
#pragma unroll
            for (int nt = 0; nt < 2; ++nt)
#pragma unroll
                for (int k = 0; k < 8; ++k)
                    pw[(k + 8 * hf) * 32 + nt * 16 + lr] = (half_t)s[nt][k];
            f16x16 pf, vfr[4];
            {
                const v8h* pp = (const v8h*)(pw + lr * 32);
                pf.h8[0] = pp[hf];
                pf.h8[1] = pp[2 + hf];
            }
#pragma unroll
            for (int nt = 0; nt < 4; ++nt) {
                const half_t* vp = &Vs[buf][nt * 16 + lr];
                for (int s = 0; s < 16; ++s) vfr[nt].v[s] = vp[(s < 8 ? 8 * hf + s : 8 + 8 * hf + s) * 64];
            }
            // O += P @ V
#pragma unroll
            for (int nt = 0; nt < 4; ++nt)
                o[nt] = __builtin_amdgcn_wmma_f32_16x16x32_f16(
                    false, pf.v, false, vfr[nt].v, (short)0, o[nt], false, false);
        }
    }

    // normalize and store per-head output (f32)
    float* Obase = Oh + (size_t)(h * B_ + b) * S_ * E_;
#pragma unroll
    for (int k = 0; k < 8; ++k) {
        float inv = 1.0f / l_r[k];
        int row = q0 + k + 8 * hf;
#pragma unroll
        for (int nt = 0; nt < 4; ++nt)
            Obase[(size_t)row * E_ + nt * 16 + lr] = o[nt][k] * inv;
    }
}

// ---------------------------------------------------------------------------
// Kernel 3: head mean + folded final projection + residual + LayerNorm
// (torch-style: unbiased std (ddof=1), divide by (std + 1e-6)).
// ---------------------------------------------------------------------------
__global__ __launch_bounds__(512) void final_kernel(
    const float* __restrict__ Oh, const float* __restrict__ Wfs,
    const float* __restrict__ bf, const float* __restrict__ query,
    const float* __restrict__ ln_w, const float* __restrict__ ln_b,
    float* __restrict__ out)
{
    __shared__ float obar[E_];
    __shared__ float red[16];

    const int row = blockIdx.x;              // 0..4095
    const int b = row >> 10, s = row & 1023;
    const int t = threadIdx.x;
    const int wave = t >> 5, lane = t & 31;

    if (t < E_) {
        float acc = 0.f;
#pragma unroll
        for (int h = 0; h < H_; ++h)
            acc += Oh[((size_t)(h * B_ + b) * S_ + s) * E_ + t];
        obar[t] = acc * (1.0f / (float)H_);
    }
    __syncthreads();

    float acc = bf[t];
#pragma unroll 8
    for (int e = 0; e < E_; ++e)
        acc = fmaf(obar[e], Wfs[e * D_ + t], acc);
    float x = acc + query[(size_t)row * D_ + t];

    // mean
    float v = x;
#pragma unroll
    for (int sh = 16; sh >= 1; sh >>= 1) v += __shfl_xor(v, sh, 32);
    if (lane == 0) red[wave] = v;
    __syncthreads();
    float tot = 0.f;
#pragma unroll
    for (int i = 0; i < 16; ++i) tot += red[i];
    const float mean = tot / (float)D_;
    __syncthreads();

    // unbiased variance
    float d = x - mean;
    float v2 = d * d;
#pragma unroll
    for (int sh = 16; sh >= 1; sh >>= 1) v2 += __shfl_xor(v2, sh, 32);
    if (lane == 0) red[wave] = v2;
    __syncthreads();
    float var = 0.f;
#pragma unroll
    for (int i = 0; i < 16; ++i) var += red[i];
    const float stdv = sqrtf(var / (float)(D_ - 1));

    out[(size_t)row * D_ + t] = ln_w[t] * d / (stdv + 1e-6f) + ln_b[t];
}

// ---------------------------------------------------------------------------
extern "C" void kernel_launch(void* const* d_in, const int* in_sizes, int n_in,
                              void* d_out, int out_size, void* d_ws, size_t ws_size,
                              hipStream_t stream) {
    (void)in_sizes; (void)n_in; (void)out_size; (void)ws_size;

    const float* query = (const float*)d_in[0];
    const float* key   = (const float*)d_in[1];
    const float* value = (const float*)d_in[2];
    /* d_in[3] = causal mask (int), implemented analytically */
    const float* Wq  = (const float*)d_in[4];
    const float* bq  = (const float*)d_in[5];
    const float* Wk  = (const float*)d_in[6];
    const float* bk  = (const float*)d_in[7];
    const float* Wv  = (const float*)d_in[8];
    const float* bv  = (const float*)d_in[9];
    const float* Wf  = (const float*)d_in[10];
    const float* bff = (const float*)d_in[11];
    const float* lnw = (const float*)d_in[12];
    const float* lnb = (const float*)d_in[13];
    float* out = (float*)d_out;

    // workspace carve (≈16.7 MiB)
    char* base = (char*)d_ws;
    half_t* Qh  = (half_t*)(base);                                   // 4 MiB
    half_t* KhT = (half_t*)(base + (size_t)4 * 1024 * 1024);         // 4 MiB [h][b][e][s]
    half_t* Vh  = (half_t*)(base + (size_t)8 * 1024 * 1024);         // 0.5 MiB
    float*  Oh  = (float*) (base + (size_t)8 * 1024 * 1024 + 512 * 1024);  // 8 MiB
    float*  Wfs = (float*) (base + (size_t)16 * 1024 * 1024 + 512 * 1024); // 128 KiB

    wfsum_kernel<<<dim3((E_ * D_ + 255) / 256), 256, 0, stream>>>(Wf, Wfs);
    proj_kernel<<<dim3(BS / 128, 2 * H_ + 1), 256, 0, stream>>>(
        query, key, value, Wq, bq, Wk, bk, Wv, bv, Qh, KhT, Vh);
    attn_kernel<<<dim3(S_ / 128, B_, H_), 256, 0, stream>>>(Qh, KhT, Vh, Oh);
    final_kernel<<<dim3(BS), 512, 0, stream>>>(Oh, Wfs, bff, query, lnw, lnb, out);
}
